// PolicyQNet_2044404432959
// MI455X (gfx1250) — matrix operand order repaired and checked by reading, not compiled, not run on hardware
//
#include <hip/hip_runtime.h>

#define N_NODES 20000
#define N_EDGES 60000
#define NODE_IN 8
#define EDGE_IN 4
#define HID 64
#define G_DIM 64
#define N_ACT 20000
#define MLP_H 128
#define BATCH 256

typedef __attribute__((ext_vector_type(16))) _Float16 v16h;
typedef __attribute__((ext_vector_type(8)))  _Float16 v8h;
typedef __attribute__((ext_vector_type(8)))  float    v8f;

// ---------------------------------------------------------------- helpers
__device__ __forceinline__ void wave_ds_fence() {
  asm volatile("s_wait_dscnt 0" ::: "memory");
}
__device__ __forceinline__ v16h v16cat(v8h lo, v8h hi) {
  v16h r;
#pragma unroll
  for (int q = 0; q < 8; ++q) { r[q] = lo[q]; r[q + 8] = hi[q]; }
  return r;
}
__device__ __forceinline__ v8h vscale(v8h v, _Float16 s) {
  v8h r;
#pragma unroll
  for (int q = 0; q < 8; ++q) r[q] = v[q] * s;
  return r;
}
__device__ __forceinline__ v8h cvt8(const float* p) {
  v8f f = *(const v8f*)p;
  v8h h;
#pragma unroll
  for (int q = 0; q < 8; ++q) h[q] = (_Float16)f[q];
  return h;
}

// ------------------------------------------------ edge-MLP hidden layer
// ehid[e,k] = relu(edge_attr[e,:] @ W1 + b1)    [60000 x 64]
__global__ void edge_hidden_kernel(const float* __restrict__ ea,
                                   const float* __restrict__ W1,
                                   const float* __restrict__ b1,
                                   float* __restrict__ out) {
  int t = blockIdx.x * blockDim.x + threadIdx.x;
  if (t >= N_EDGES * HID) return;
  int e = t >> 6, k = t & 63;
  float s = b1[k];
#pragma unroll
  for (int j = 0; j < EDGE_IN; ++j) s += ea[e * EDGE_IN + j] * W1[j * HID + k];
  out[t] = fmaxf(s, 0.0f);
}

// ------------------------------------------------ B-matrix packing (WMMA lane layout)
// Packed element for logical B'[r,o]: s=r>>5, hi=(r>>4)&1, j=r&15,
// lane=(o&15)|(hi<<4), tile=o>>4, pos=((s*NT+tile)*32+lane)*16+j
__global__ void pack_edge_B_kernel(const float* __restrict__ W2,
                                   const float* __restrict__ b2,
                                   _Float16* __restrict__ out,
                                   int Ktot, int LOG_IN) {
  int tid = blockIdx.x * blockDim.x + threadIdx.x;
  if (tid >= Ktot * HID) return;
  int r = tid >> 6, o = tid & 63;
  int IN = 1 << LOG_IN;
  int k = r >> LOG_IN, i = r & (IN - 1);
  float val;
  if (k < HID)       val = W2[(k << LOG_IN) * HID + i * HID + o];
  else if (k == HID) val = b2[i * HID + o];
  else               val = 0.0f;
  int s = r >> 5, rem = r & 31, hi = rem >> 4, j = rem & 15;
  int lane = (o & 15) | (hi << 4), t = o >> 4;
  out[((s * 4 + t) * 32 + lane) * 16 + j] = (_Float16)val;
}

__global__ void pack_rowmajor_B_kernel(const float* __restrict__ W,
                                       _Float16* __restrict__ out,
                                       int K, int OUT) {
  int tid = blockIdx.x * blockDim.x + threadIdx.x;
  if (tid >= K * OUT) return;
  int r = tid / OUT, o = tid % OUT;
  float val = W[tid];
  int NT = OUT >> 4;
  int s = r >> 5, rem = r & 31, hi = rem >> 4, j = rem & 15;
  int lane = (o & 15) | (hi << 4), t = o >> 4;
  out[((s * NT + t) * 32 + lane) * 16 + j] = (_Float16)val;
}

// ------------------------------------------------ degree
__global__ void degree_kernel(const int* __restrict__ dst, float* __restrict__ deg) {
  int t = blockIdx.x * blockDim.x + threadIdx.x;
  if (t >= N_EDGES) return;
  atomicAdd(&deg[dst[t]], 1.0f);
}

// ------------------------------------------------ NNConv message GEMM
// One wave per 16-edge tile. A[e, r=(k,i)] = h_aug[e,k] * v[e,i] built on the fly,
// K = (HID+1)*IN padded to multiple of 32. D scatter-added into agg[dst[e], :].
template <int IN, int LOG_IN, int KSTEPS>
__global__ __launch_bounds__(32) void nnconv_gemm_kernel(
    const float* __restrict__ vfeat, const float* __restrict__ ehid,
    const _Float16* __restrict__ Bpack, const int* __restrict__ src,
    const int* __restrict__ dst, float* __restrict__ agg) {
  __shared__ alignas(32) _Float16 h_s[16 * 72];   // k slots 0..67 (aug + zero pad)
  __shared__ alignas(32) _Float16 v_s[16 * IN];
  __shared__ int src_s[16];
  __shared__ int dst_s[16];

  const int lane = threadIdx.x;
  const int e0 = blockIdx.x * 16;

  if (lane < 16) {
    src_s[lane] = src[e0 + lane];
    dst_s[lane] = dst[e0 + lane];
  }
  wave_ds_fence();

  // stage gathered source-node features as f16
  for (int idx = lane; idx < 16 * IN; idx += 32) {
    int row = idx >> LOG_IN;
    int i = idx & (IN - 1);
    v_s[idx] = (_Float16)vfeat[src_s[row] * IN + i];
  }
  // stage edge-hidden row (augmented with 1.0, zero padded)
  {
    int row = lane & 15;
    int e = e0 + row;
    int k0 = (lane >> 4) * 34;
    for (int kk = 0; kk < 34; ++kk) {
      int k = k0 + kk;
      float v = (k < HID) ? ehid[e * HID + k] : ((k == HID) ? 1.0f : 0.0f);
      h_s[row * 72 + k] = (_Float16)v;
    }
  }
  wave_ds_fence();

  v8f acc[4] = {};
  const int row = lane & 15;
  const int hiHalf = lane >> 4;
  const v16h* bp = (const v16h*)Bpack + lane;

  for (int s = 0; s < KSTEPS; ++s) {
    const int base = s * 32 + hiHalf * 16;
    const int rA = base, rB = base + 8;
    _Float16 hA = h_s[row * 72 + (rA >> LOG_IN)];
    _Float16 hB = h_s[row * 72 + (rB >> LOG_IN)];
    v8h vA = *(const v8h*)&v_s[row * IN + (rA & (IN - 1))];
    v8h vB = *(const v8h*)&v_s[row * IN + (rB & (IN - 1))];
    v16h A = v16cat(vscale(vA, hA), vscale(vB, hB));
    if (s + 1 < KSTEPS)
      __builtin_prefetch((const void*)(bp + (s + 1) * 128), 0, 1);
#pragma unroll
    for (int t = 0; t < 4; ++t) {
      v16h Bt = bp[(s * 4 + t) * 32];
      acc[t] = __builtin_amdgcn_wmma_f32_16x16x32_f16(false, A, false, Bt,
                                                      (short)0, acc[t], false, false);
    }
  }

  // scatter-add: lane holds N=lane&15, rows M = hiHalf*8 + j
  const int n = lane & 15;
#pragma unroll
  for (int t = 0; t < 4; ++t) {
    int o = t * 16 + n;
#pragma unroll
    for (int j = 0; j < 8; ++j) {
      int m = hiHalf * 8 + j;
      atomicAdd(&agg[dst_s[m] * HID + o], acc[t][j]);
    }
  }
}

// ------------------------------------------------ node update (+ optional pool)
__global__ void node_update_kernel(const float* __restrict__ vin,
                                   const float* __restrict__ agg,
                                   const float* __restrict__ deg,
                                   const float* __restrict__ root,
                                   const float* __restrict__ bias,
                                   float* __restrict__ out, int IN, int doPool,
                                   float* __restrict__ pool) {
  int t = blockIdx.x * blockDim.x + threadIdx.x;
  if (t >= N_NODES * HID) return;
  int n = t >> 6, o = t & 63;
  float s = agg[t] / fmaxf(deg[n], 1.0f) + bias[o];
  for (int i = 0; i < IN; ++i) s += vin[n * IN + i] * root[i * HID + o];
  s = fmaxf(s, 0.0f);
  out[t] = s;
  if (doPool) atomicAdd(&pool[o], s);
}

// ------------------------------------------------ global pool + proj
__global__ void proj_kernel(const float* __restrict__ pool,
                            const float* __restrict__ projW,
                            const float* __restrict__ projb,
                            float* __restrict__ g) {
  int o = threadIdx.x;
  if (o >= G_DIM) return;
  float s = projb[o];
  for (int h = 0; h < HID; ++h)
    s += (pool[h] * (1.0f / (float)N_NODES)) * projW[h * G_DIM + o];
  g[o] = s;
}

// ------------------------------------------------ MLP layer1 GEMM (split-K WMMA)
// z1acc[b,o] += cat(a,g)[b,:] @ mW1 ; K = 20064 = 627*32, a-region = chunks 0..624
__global__ __launch_bounds__(32) void mlp1_gemm_kernel(
    const float* __restrict__ a, const float* __restrict__ g,
    const _Float16* __restrict__ Bpack, float* __restrict__ z1acc) {
  const int lane = threadIdx.x;
  const int mtile = blockIdx.x;              // 0..15
  const int s0 = blockIdx.y * 19;            // 33 chunks * 19 = 627
  const int s1 = (s0 + 19 < 627) ? s0 + 19 : 627;
  const int row = lane & 15;
  const int hiHalf = lane >> 4;
  const int brow = mtile * 16 + row;
  const float* arow = a + (size_t)brow * N_ACT;
  const v16h* bp = (const v16h*)Bpack + lane;

  v8f acc[8] = {};
  for (int s = s0; s < s1; ++s) {
    const int base = s * 32 + hiHalf * 16;
    const float* pA = (base < N_ACT) ? (arow + base) : (g + (base - N_ACT));
    const float* pB = (base + 8 < N_ACT) ? (arow + base + 8) : (g + (base + 8 - N_ACT));
    v16h A = v16cat(cvt8(pA), cvt8(pB));
    if (s + 1 < s1)
      __builtin_prefetch((const void*)(bp + (s + 1) * 256), 0, 1);
#pragma unroll
    for (int t = 0; t < 8; ++t) {
      v16h Bt = bp[(s * 8 + t) * 32];
      acc[t] = __builtin_amdgcn_wmma_f32_16x16x32_f16(false, A, false, Bt,
                                                      (short)0, acc[t], false, false);
    }
  }
  const int n = lane & 15;
#pragma unroll
  for (int t = 0; t < 8; ++t) {
    int o = t * 16 + n;
#pragma unroll
    for (int j = 0; j < 8; ++j) {
      int m = hiHalf * 8 + j;
      atomicAdd(&z1acc[(mtile * 16 + m) * MLP_H + o], acc[t][j]);
    }
  }
}

// ------------------------------------------------ MLP tails
__global__ void bias_relu_kernel(float* __restrict__ z, const float* __restrict__ b) {
  int t = blockIdx.x * blockDim.x + threadIdx.x;
  if (t >= BATCH * MLP_H) return;
  z[t] = fmaxf(z[t] + b[t & 127], 0.0f);
}
__global__ void mlp2_kernel(const float* __restrict__ z1, const float* __restrict__ W,
                            const float* __restrict__ b, float* __restrict__ z2) {
  int t = blockIdx.x * blockDim.x + threadIdx.x;
  if (t >= BATCH * MLP_H) return;
  int bb = t >> 7, o = t & 127;
  float s = b[o];
  for (int h = 0; h < MLP_H; ++h) s += z1[bb * MLP_H + h] * W[h * MLP_H + o];
  z2[t] = fmaxf(s, 0.0f);
}
__global__ void mlp3_kernel(const float* __restrict__ z2, const float* __restrict__ W,
                            const float* __restrict__ b, float* __restrict__ out) {
  int t = blockIdx.x * blockDim.x + threadIdx.x;
  if (t >= BATCH) return;
  float s = b[0];
  for (int h = 0; h < MLP_H; ++h) s += z2[t * MLP_H + h] * W[h];
  out[t] = s;
}

// ================================================================= launch
extern "C" void kernel_launch(void* const* d_in, const int* in_sizes, int n_in,
                              void* d_out, int out_size, void* d_ws, size_t ws_size,
                              hipStream_t stream) {
  const float* x      = (const float*)d_in[0];
  const int*   ei     = (const int*)d_in[1];
  const float* ea     = (const float*)d_in[2];
  const float* a      = (const float*)d_in[3];
  const float* e1W1   = (const float*)d_in[4];
  const float* e1b1   = (const float*)d_in[5];
  const float* e1W2   = (const float*)d_in[6];
  const float* e1b2   = (const float*)d_in[7];
  const float* root1  = (const float*)d_in[8];
  const float* bias1  = (const float*)d_in[9];
  const float* e2W1   = (const float*)d_in[10];
  const float* e2b1   = (const float*)d_in[11];
  const float* e2W2   = (const float*)d_in[12];
  const float* e2b2   = (const float*)d_in[13];
  const float* root2  = (const float*)d_in[14];
  const float* bias2  = (const float*)d_in[15];
  const float* projW  = (const float*)d_in[16];
  const float* projb  = (const float*)d_in[17];
  const float* mW1    = (const float*)d_in[18];
  const float* mb1    = (const float*)d_in[19];
  const float* mW2    = (const float*)d_in[20];
  const float* mb2    = (const float*)d_in[21];
  const float* mW3    = (const float*)d_in[22];
  const float* mb3    = (const float*)d_in[23];
  const int* src = ei;
  const int* dst = ei + N_EDGES;

  // workspace carve-up (256B aligned)
  char* w = (char*)d_ws;
  size_t off = 0;
  auto carve = [&](size_t bytes) -> void* {
    void* p = (void*)(w + off);
    off = (off + bytes + 255) & ~(size_t)255;
    return p;
  };
  float*    ehid   = (float*)carve((size_t)N_EDGES * HID * 4);       // 15.36 MB
  float*    agg    = (float*)carve((size_t)N_NODES * HID * 4);       // 5.12 MB
  float*    deg    = (float*)carve((size_t)N_NODES * 4);
  float*    h1     = (float*)carve((size_t)N_NODES * HID * 4);
  float*    h2     = (float*)carve((size_t)N_NODES * HID * 4);
  _Float16* b1p    = (_Float16*)carve((size_t)544 * HID * 2);
  _Float16* b2p    = (_Float16*)carve((size_t)4160 * HID * 2);
  _Float16* mW1p   = (_Float16*)carve((size_t)(N_ACT + G_DIM) * MLP_H * 2);
  float*    z1acc  = (float*)carve((size_t)BATCH * MLP_H * 4);
  float*    z2     = (float*)carve((size_t)BATCH * MLP_H * 4);
  float*    pool   = (float*)carve(256);
  float*    g      = (float*)carve(256);
  (void)ws_size; (void)n_in; (void)in_sizes; (void)out_size;

  hipMemsetAsync(agg, 0, (size_t)N_NODES * HID * 4, stream);
  hipMemsetAsync(deg, 0, (size_t)N_NODES * 4, stream);
  hipMemsetAsync(pool, 0, G_DIM * 4, stream);
  hipMemsetAsync(z1acc, 0, (size_t)BATCH * MLP_H * 4, stream);

  // pack B matrices into WMMA lane layout
  pack_edge_B_kernel<<<(544 * HID + 255) / 256, 256, 0, stream>>>(e1W2, e1b2, b1p, 544, 3);
  pack_edge_B_kernel<<<(4160 * HID + 255) / 256, 256, 0, stream>>>(e2W2, e2b2, b2p, 4160, 6);
  pack_rowmajor_B_kernel<<<((N_ACT + G_DIM) * MLP_H + 255) / 256, 256, 0, stream>>>(
      mW1, mW1p, N_ACT + G_DIM, MLP_H);

  degree_kernel<<<(N_EDGES + 255) / 256, 256, 0, stream>>>(dst, deg);

  // ---- NNConv layer 1 ----
  edge_hidden_kernel<<<(N_EDGES * HID + 255) / 256, 256, 0, stream>>>(ea, e1W1, e1b1, ehid);
  nnconv_gemm_kernel<NODE_IN, 3, 17><<<N_EDGES / 16, 32, 0, stream>>>(x, ehid, b1p, src, dst, agg);
  node_update_kernel<<<(N_NODES * HID + 255) / 256, 256, 0, stream>>>(
      x, agg, deg, root1, bias1, h1, NODE_IN, 0, pool);

  // ---- NNConv layer 2 ----
  hipMemsetAsync(agg, 0, (size_t)N_NODES * HID * 4, stream);
  edge_hidden_kernel<<<(N_EDGES * HID + 255) / 256, 256, 0, stream>>>(ea, e2W1, e2b1, ehid);
  nnconv_gemm_kernel<HID, 6, 130><<<N_EDGES / 16, 32, 0, stream>>>(h1, ehid, b2p, src, dst, agg);
  node_update_kernel<<<(N_NODES * HID + 255) / 256, 256, 0, stream>>>(
      h1, agg, deg, root2, bias2, h2, HID, 1, pool);

  // ---- pool + proj ----
  proj_kernel<<<1, 64, 0, stream>>>(pool, projW, projb, g);

  // ---- ActionQMLP ----
  dim3 mg(16, 33);
  mlp1_gemm_kernel<<<mg, 32, 0, stream>>>(a, g, mW1p, z1acc);
  bias_relu_kernel<<<(BATCH * MLP_H + 255) / 256, 256, 0, stream>>>(z1acc, mb1);
  mlp2_kernel<<<(BATCH * MLP_H + 255) / 256, 256, 0, stream>>>(z1acc, mW2, mb2, z2);
  mlp3_kernel<<<(BATCH + 255) / 256, 256, 0, stream>>>(z2, mW3, mb3, (float*)d_out);
}
